// LLM_17248588661292
// MI455X (gfx1250) — compile-verified, matrix operand order checked by reading, not run on hardware
//
#include <hip/hip_runtime.h>
#include <hip/hip_bf16.h>

// ---------------------------------------------------------------------------
// CDNA5 / gfx1250 LLaMA-forward. wave32; all GEMMs via v_wmma_f32_16x16x32_bf16
// B-operand double-buffered through LDS via GLOBAL_LOAD_ASYNC_TO_LDS_B128,
// bf16 packing via native fptrunc (v_cvt_pk_bf16_f32 where available).
// ---------------------------------------------------------------------------

typedef __attribute__((ext_vector_type(16))) __bf16 v16bf;
typedef __attribute__((ext_vector_type(2)))  __bf16 v2bf;
typedef __attribute__((ext_vector_type(8)))  float  v8f;
typedef __attribute__((ext_vector_type(4)))  int    v4i;

typedef __attribute__((address_space(1))) v4i gv4i;   // global v4i
typedef __attribute__((address_space(3))) v4i lv4i;   // LDS v4i

#define HAS_ASYNC_LDS  __has_builtin(__builtin_amdgcn_global_load_async_to_lds_b128)
#define HAS_ASYNC_WAIT __has_builtin(__builtin_amdgcn_s_wait_asynccnt)
#define HAS_PK_BF16    __has_builtin(__builtin_amdgcn_cvt_pk_bf16_f32)

union ABf16 {
    v16bf        v;
    unsigned int u[8];
};

__device__ __forceinline__ unsigned pack2(float lo, float hi) {
    union { v2bf b; unsigned u; } c;
#if HAS_PK_BF16
    c.b = __builtin_amdgcn_cvt_pk_bf16_f32(lo, hi);
#else
    v2bf t;
    t[0] = (__bf16)lo;                 // native fptrunc f32->bf16 (RNE)
    t[1] = (__bf16)hi;
    c.b = t;
#endif
    return c.u;
}
__device__ __forceinline__ unsigned short f2bf(float f) {
    union { __bf16 b; unsigned short s; } c;
    c.b = (__bf16)f;
    return c.s;
}

__device__ __forceinline__ float wave_red_sum(float v) {
    for (int off = 16; off; off >>= 1) v += __shfl_xor(v, off, 32);
    return v;
}
__device__ __forceinline__ float wave_red_max(float v) {
    for (int off = 16; off; off >>= 1) v = fmaxf(v, __shfl_xor(v, off, 32));
    return v;
}

// ---------------------------------------------------------------------------
// Embedding gather: h[row, c] = emb[tok[row], c]   (H = 1024)
// ---------------------------------------------------------------------------
__global__ void embed_kernel(const int* __restrict__ tok,
                             const float* __restrict__ emb,
                             float* __restrict__ h) {
    int i   = blockIdx.x * blockDim.x + threadIdx.x;
    int col = i & 1023;
    int row = i >> 10;
    h[i] = emb[(size_t)tok[row] * 1024 + col];
}

// ---------------------------------------------------------------------------
// RMSNorm over H=1024, output bf16 (feeds WMMA A-operand). One block per row.
// ---------------------------------------------------------------------------
__global__ __launch_bounds__(256) void rmsnorm_bf16(const float* __restrict__ x,
                                                    const float* __restrict__ w,
                                                    unsigned short* __restrict__ o) {
    int row = blockIdx.x, tid = threadIdx.x;
    const float* xr = x + (size_t)row * 1024;
    float ss = 0.f;
#pragma unroll
    for (int c = tid; c < 1024; c += 256) { float v = xr[c]; ss += v * v; }
    ss = wave_red_sum(ss);
    __shared__ float red[8];
    if ((tid & 31) == 0) red[tid >> 5] = ss;
    __syncthreads();
    float tot = 0.f;
#pragma unroll
    for (int i = 0; i < 8; ++i) tot += red[i];
    float inv = rsqrtf(tot * (1.0f / 1024.0f) + 1e-6f);
#pragma unroll
    for (int c = tid; c < 1024; c += 256)
        o[(size_t)row * 1024 + c] = f2bf(xr[c] * inv * w[c]);
}

// ---------------------------------------------------------------------------
// GEMM: C[M,N] = A(bf16)[M,K] @ B(fp32,row-major)[K,N] (+ resid)
// Block = 16(M) x 128(N) strip: 8 waves, one 16x16 WMMA tile each, sharing
// the A tile. B tiles (32x128 fp32) are double-buffered through LDS with
// async B128 copies overlapping WMMA compute; packed to bf16 wave32 layout:
//   lanes 0-15 : N=lane,    K = 0..7 (v0-3) and 16..23 (v4-7)
//   lanes 16-31: N=lane-16, K = 8..15 and 24..31
// ---------------------------------------------------------------------------
__global__ __launch_bounds__(256) void gemm_bf16(const unsigned short* __restrict__ A,
                                                 const float* __restrict__ B,
                                                 const float* __restrict__ resid,
                                                 float* __restrict__ C,
                                                 int M, int N, int K) {
    __shared__ float sB[2][32 * 128];  // 2 x 16 KB B-tile double buffer
    int tid  = threadIdx.x;
    int lane = tid & 31;
    int wave = tid >> 5;
    int nstrips = N >> 7;              // N / 128
    int tm = blockIdx.x / nstrips;
    int ns = blockIdx.x - tm * nstrips;
    int colBase = ns << 7;

    int mrow = lane & 15;
    int kb0  = (lane >> 4) << 3;       // 0 or 8
    int colLocal = (wave << 4) + mrow; // this wave's 16-col sub-tile

    auto stage = [&](float* dst, int k0) {
#pragma unroll
        for (int j = 0; j < 4; ++j) {
            int linear = tid + 256 * j;          // 0..1023 16B-transfers
            int kr = linear >> 5;                // 0..31
            int ci = (linear & 31) << 2;         // 0,4,...,124
            const float* gsrc = B + (size_t)(k0 + kr) * N + colBase + ci;
            float*       ldst = dst + kr * 128 + ci;
#if HAS_ASYNC_LDS
            __builtin_amdgcn_global_load_async_to_lds_b128(
                (gv4i*)(uintptr_t)gsrc, (lv4i*)(unsigned)(uintptr_t)ldst, 0, 0);
#else
            *(float4*)ldst = *(const float4*)gsrc;
#endif
        }
    };

    const unsigned* arow = (const unsigned*)(A + (size_t)((tm << 4) + mrow) * K);
    v8f acc = {};
    stage(&sB[0][0], 0);               // prologue: first tile in flight
    int cur = 0;
    for (int k0 = 0; k0 < K; k0 += 32) {
        if (k0 + 32 < K) {
            stage(&sB[cur ^ 1][0], k0 + 32);     // next tile in flight
#if HAS_ASYNC_LDS && HAS_ASYNC_WAIT
            __builtin_amdgcn_s_wait_asynccnt(4); // drain previous group only
#endif
        } else {
#if HAS_ASYNC_LDS && HAS_ASYNC_WAIT
            __builtin_amdgcn_s_wait_asynccnt(0);
#endif
        }
        __syncthreads();               // all waves' copies of sB[cur] landed

        ABf16 a, bm;
#pragma unroll
        for (int r = 0; r < 4; ++r) {
            a.u[r]     = arow[(k0 + kb0 + 2 * r) >> 1];
            a.u[r + 4] = arow[(k0 + 16 + kb0 + 2 * r) >> 1];
        }
        const float* sb = &sB[cur][0];
#pragma unroll
        for (int r = 0; r < 4; ++r) {
            int k = kb0 + 2 * r;
            bm.u[r]     = pack2(sb[k * 128 + colLocal],        sb[(k + 1) * 128 + colLocal]);
            bm.u[r + 4] = pack2(sb[(k + 16) * 128 + colLocal], sb[(k + 17) * 128 + colLocal]);
        }
        acc = __builtin_amdgcn_wmma_f32_16x16x32_bf16(false, a.v, false, bm.v,
                                                      (short)0, acc, false, false);
        __syncthreads();               // protect sB[cur] before re-stage
        cur ^= 1;
    }
    int col   = colBase + colLocal;
    int mbase = (tm << 4) + ((lane >> 4) << 3);
    if (resid) {
#pragma unroll
        for (int i = 0; i < 8; ++i) {
            size_t idx = (size_t)(mbase + i) * N + col;
            C[idx] = acc[i] + resid[idx];
        }
    } else {
#pragma unroll
        for (int i = 0; i < 8; ++i) {
            size_t idx = (size_t)(mbase + i) * N + col;
            C[idx] = acc[i];
        }
    }
}

// ---------------------------------------------------------------------------
// RoPE (interleaved) + convert to bf16.  x: [B,S,nh,64] fp32.  total = B*S*nh*32
// ---------------------------------------------------------------------------
__global__ void rope_bf16(const float* __restrict__ x,
                          const float* __restrict__ cs,
                          const float* __restrict__ sn,
                          unsigned short* __restrict__ o,
                          int nh, int total) {
    int idx = blockIdx.x * blockDim.x + threadIdx.x;
    if (idx >= total) return;
    int d2 = idx & 31;
    int t  = idx >> 5;                 // (b*S + s)*nh + head
    int s  = (t / nh) & 1023;          // S = 1024
    size_t base = (size_t)t * 64;
    float xr = x[base + 2 * d2], xi = x[base + 2 * d2 + 1];
    float c = cs[s * 32 + d2], si = sn[s * 32 + d2];
    unsigned pr = pack2(xr * c - xi * si, xr * si + xi * c);
    *(unsigned*)(o + base + 2 * d2) = pr;        // 2*d2 even -> 4B aligned
}

__global__ void conv_bf16(const float* __restrict__ x, unsigned short* __restrict__ o, int n) {
    int i = blockIdx.x * blockDim.x + threadIdx.x;
    if (i < n) o[i] = f2bf(x[i]);
}

__global__ void silu_mul_bf16(const float* __restrict__ g, const float* __restrict__ u,
                              unsigned short* __restrict__ o, int n) {
    int i = blockIdx.x * blockDim.x + threadIdx.x;
    if (i < n) {
        float gv = g[i];
        float sv = gv / (1.f + __expf(-gv));
        o[i] = f2bf(sv * u[i]);
    }
}

// ---------------------------------------------------------------------------
// Fused causal attention for one (b, head, 16-row q-tile).
// Phase 1: scores[16,1024] = q_tile @ K^T / 8 (+mask) into LDS, WMMA K=64.
// Phase 2: row softmax in LDS (wave32 shfl reductions).
// Phase 3: out[16,64] = P @ V; all 8 waves: d-tile = wave&3, K-half = wave>>2,
//          partials combined through 4 KB LDS. grid = B*HQ*(S/16) = 2048.
// ---------------------------------------------------------------------------
__global__ __launch_bounds__(256) void attn_kernel(const unsigned short* __restrict__ qb,
                                                   const unsigned short* __restrict__ kbf,
                                                   const unsigned short* __restrict__ vb,
                                                   float* __restrict__ out) {
    __shared__ float sc[16][1024];      // 64 KB scores
    __shared__ float sacc[16][64];      // 4 KB P@V partial combine
    int qt  = blockIdx.x & 63;
    int bh  = blockIdx.x >> 6;
    int b   = bh >> 4;
    int h   = bh & 15;
    int kvh = h >> 2;                   // G = 4
    int tid = threadIdx.x;
    int lane = tid & 31, wave = tid >> 5;
    int mrow = lane & 15;
    int kb0  = (lane >> 4) << 3;
    int m0   = (lane >> 4) << 3;

    // Preload q-tile A operand (D = 64 -> two K=32 slabs)
    const unsigned* qrow =
        (const unsigned*)(qb + (((size_t)b * 1024 + qt * 16 + mrow) * 16 + h) * 64);
    ABf16 a0, a1;
#pragma unroll
    for (int r = 0; r < 4; ++r) {
        a0.u[r]     = qrow[(kb0 + 2 * r) >> 1];
        a0.u[r + 4] = qrow[(16 + kb0 + 2 * r) >> 1];
        a1.u[r]     = qrow[(32 + kb0 + 2 * r) >> 1];
        a1.u[r + 4] = qrow[(48 + kb0 + 2 * r) >> 1];
    }

    // ---- Phase 1: scores ----
    for (int kt = wave; kt < 64; kt += 8) {
        if (kt <= qt) {
            const unsigned* krow =
                (const unsigned*)(kbf + (((size_t)b * 1024 + kt * 16 + mrow) * 4 + kvh) * 64);
            ABf16 b0, b1;
#pragma unroll
            for (int r = 0; r < 4; ++r) {
                b0.u[r]     = krow[(kb0 + 2 * r) >> 1];
                b0.u[r + 4] = krow[(16 + kb0 + 2 * r) >> 1];
                b1.u[r]     = krow[(32 + kb0 + 2 * r) >> 1];
                b1.u[r + 4] = krow[(48 + kb0 + 2 * r) >> 1];
            }
            v8f acc = {};
            acc = __builtin_amdgcn_wmma_f32_16x16x32_bf16(false, a0.v, false, b0.v,
                                                          (short)0, acc, false, false);
            acc = __builtin_amdgcn_wmma_f32_16x16x32_bf16(false, a1.v, false, b1.v,
                                                          (short)0, acc, false, false);
#pragma unroll
            for (int i = 0; i < 8; ++i) {
                int m  = m0 + i;
                int qi = qt * 16 + m;
                int ki = kt * 16 + mrow;
                sc[m][kt * 16 + mrow] = (ki <= qi) ? acc[i] * 0.125f : -__builtin_inff();
            }
        } else {
#pragma unroll
            for (int i = 0; i < 8; ++i) sc[m0 + i][kt * 16 + mrow] = -__builtin_inff();
        }
    }
    __syncthreads();

    // ---- Phase 2: softmax (each wave owns 2 rows) ----
#pragma unroll
    for (int rr = 0; rr < 2; ++rr) {
        int r = wave * 2 + rr;
        float mx = -__builtin_inff();
        for (int c = lane; c < 1024; c += 32) mx = fmaxf(mx, sc[r][c]);
        mx = wave_red_max(mx);
        float sum = 0.f;
        for (int c = lane; c < 1024; c += 32) {
            float e = __expf(sc[r][c] - mx);
            sc[r][c] = e;
            sum += e;
        }
        sum = wave_red_sum(sum);
        float inv = 1.f / sum;
        for (int c = lane; c < 1024; c += 32) sc[r][c] *= inv;
    }
    __syncthreads();

    // ---- Phase 3: out = P @ V, all 8 waves ----
    {
        int nt = wave & 3;              // d-tile
        int kh = wave >> 2;             // K half
        int d  = nt * 16 + mrow;
        v8f acc = {};
        int kend = kh * 512 + 512;
        for (int k0 = kh * 512; k0 < kend; k0 += 32) {
            ABf16 a, bm;
#pragma unroll
            for (int r = 0; r < 4; ++r) {
                int k = k0 + kb0 + 2 * r;
                a.u[r]     = pack2(sc[mrow][k], sc[mrow][k + 1]);
                a.u[r + 4] = pack2(sc[mrow][k + 16], sc[mrow][k + 17]);
            }
#pragma unroll
            for (int r = 0; r < 4; ++r) {
                int k = k0 + kb0 + 2 * r;
                size_t i0 = (((size_t)b * 1024 + k) * 4 + kvh) * 64 + d;
                size_t i1 = (((size_t)b * 1024 + k + 1) * 4 + kvh) * 64 + d;
                bm.u[r] = (unsigned)vb[i0] | ((unsigned)vb[i1] << 16);
                size_t j0 = (((size_t)b * 1024 + k + 16) * 4 + kvh) * 64 + d;
                size_t j1 = (((size_t)b * 1024 + k + 17) * 4 + kvh) * 64 + d;
                bm.u[r + 4] = (unsigned)vb[j0] | ((unsigned)vb[j1] << 16);
            }
            acc = __builtin_amdgcn_wmma_f32_16x16x32_bf16(false, a.v, false, bm.v,
                                                          (short)0, acc, false, false);
        }
        if (wave >= 4) {
#pragma unroll
            for (int i = 0; i < 8; ++i) sacc[m0 + i][d] = acc[i];
        }
        __syncthreads();
        if (wave < 4) {
#pragma unroll
            for (int i = 0; i < 8; ++i) {
                int m = m0 + i;
                out[((size_t)b * 1024 + qt * 16 + m) * 1024 + h * 64 + d] =
                    acc[i] + sacc[m][d];
            }
        }
    }
}

// ---------------------------------------------------------------------------
// Host-side orchestration
// ---------------------------------------------------------------------------
static void launch_gemm(const unsigned short* A, const float* B, const float* resid,
                        float* C, int M, int N, int K, hipStream_t s) {
    int blocks = (M / 16) * (N / 128);
    gemm_bf16<<<blocks, 256, 0, s>>>(A, B, resid, C, M, N, K);
}

extern "C" void kernel_launch(void* const* d_in, const int* in_sizes, int n_in,
                              void* d_out, int out_size, void* d_ws, size_t ws_size,
                              hipStream_t stream) {
    (void)in_sizes; (void)n_in; (void)out_size; (void)ws_size;
    const int*   tokens  = (const int*)d_in[0];
    const float* tok_emb = (const float*)d_in[1];
    const float* wq = (const float*)d_in[2];
    const float* wk = (const float*)d_in[3];
    const float* wv = (const float*)d_in[4];
    const float* wo = (const float*)d_in[5];
    const float* w1 = (const float*)d_in[6];
    const float* w2 = (const float*)d_in[7];
    const float* w3 = (const float*)d_in[8];
    const float* an = (const float*)d_in[9];
    const float* fn = (const float*)d_in[10];
    const float* nw = (const float*)d_in[11];
    const float* ow = (const float*)d_in[12];
    const float* cs = (const float*)d_in[13];
    const float* sn = (const float*)d_in[14];

    const int BS = 2048, H = 1024, Ii = 2816, L = 8, Vv = 32000;

    char* p = (char*)d_ws;
    auto alloc = [&](size_t bytes) -> void* {
        void* r = (void*)p;
        p += (bytes + 255) & ~(size_t)255;
        return r;
    };
    float*          h    = (float*)alloc((size_t)BS * H * 4);
    unsigned short* xb   = (unsigned short*)alloc((size_t)BS * H * 2);
    float*          qf   = (float*)alloc((size_t)BS * H * 4);
    float*          kf   = (float*)alloc((size_t)BS * 256 * 4);
    float*          vf   = (float*)alloc((size_t)BS * 256 * 4);
    unsigned short* qbuf = (unsigned short*)alloc((size_t)BS * H * 2);
    unsigned short* kbuf = (unsigned short*)alloc((size_t)BS * 256 * 2);
    unsigned short* vbuf = (unsigned short*)alloc((size_t)BS * 256 * 2);
    float*          ao   = (float*)alloc((size_t)BS * H * 4);
    unsigned short* aob  = (unsigned short*)alloc((size_t)BS * H * 2);
    float*          gbuf = (float*)alloc((size_t)BS * Ii * 4);
    float*          ubuf = (float*)alloc((size_t)BS * Ii * 4);
    unsigned short* sbuf = (unsigned short*)alloc((size_t)BS * Ii * 2);

    embed_kernel<<<(BS * H) / 256, 256, 0, stream>>>(tokens, tok_emb, h);

    for (int l = 0; l < L; ++l) {
        const float* wq_l = wq + (size_t)l * H * 1024;
        const float* wk_l = wk + (size_t)l * H * 256;
        const float* wv_l = wv + (size_t)l * H * 256;
        const float* wo_l = wo + (size_t)l * 1024 * H;
        const float* w1_l = w1 + (size_t)l * H * Ii;
        const float* w2_l = w2 + (size_t)l * Ii * H;
        const float* w3_l = w3 + (size_t)l * H * Ii;

        rmsnorm_bf16<<<BS, 256, 0, stream>>>(h, an + (size_t)l * H, xb);
        launch_gemm(xb, wq_l, nullptr, qf, BS, 1024, H, stream);
        launch_gemm(xb, wk_l, nullptr, kf, BS, 256, H, stream);
        launch_gemm(xb, wv_l, nullptr, vf, BS, 256, H, stream);

        rope_bf16<<<(BS * 16 * 32) / 256, 256, 0, stream>>>(qf, cs, sn, qbuf, 16, BS * 16 * 32);
        rope_bf16<<<(BS * 4 * 32) / 256, 256, 0, stream>>>(kf, cs, sn, kbuf, 4, BS * 4 * 32);
        conv_bf16<<<(BS * 256) / 256, 256, 0, stream>>>(vf, vbuf, BS * 256);

        attn_kernel<<<2048, 256, 0, stream>>>(qbuf, kbuf, vbuf, ao);

        conv_bf16<<<(BS * H) / 256, 256, 0, stream>>>(ao, aob, BS * H);
        launch_gemm(aob, wo_l, h, h, BS, H, 1024, stream);

        rmsnorm_bf16<<<BS, 256, 0, stream>>>(h, fn + (size_t)l * H, xb);
        launch_gemm(xb, w1_l, nullptr, gbuf, BS, Ii, H, stream);
        launch_gemm(xb, w3_l, nullptr, ubuf, BS, Ii, H, stream);
        silu_mul_bf16<<<(BS * Ii) / 256, 256, 0, stream>>>(gbuf, ubuf, sbuf, BS * Ii);
        launch_gemm(sbuf, w2_l, h, h, BS, H, Ii, stream);
    }

    rmsnorm_bf16<<<BS, 256, 0, stream>>>(h, nw, xb);
    launch_gemm(xb, ow, nullptr, (float*)d_out, BS, Vv, H, stream);
}
